// TransformerBlock_88682484728068
// MI455X (gfx1250) — matrix operand order repaired and checked by reading, not compiled, not run on hardware
//
#include <hip/hip_runtime.h>
#include <math.h>

typedef __attribute__((ext_vector_type(16))) _Float16 v16h;
typedef __attribute__((ext_vector_type(8)))  _Float16 v8h;
typedef __attribute__((ext_vector_type(8)))  float    v8f;

union V16 { v16h v; v8h h[2]; };

#define EMBED  1024
#define HEADS  16
#define HDIM   64
#define HIDDEN 4096
#define SEQ    2048
#define TOKENS 4096   // B*N = 2*2048

// ---------------------------------------------------------------------------
// CDNA5 async-copy helpers: global -> LDS DMA tracked by ASYNCcnt.
// VDST carries a per-lane LDS byte address (generic -> AS(3) cast).
// ---------------------------------------------------------------------------
typedef __attribute__((address_space(3))) void lds_void;

__device__ inline unsigned lds_addr(void* p) {
  return (unsigned)(unsigned long long)(lds_void*)p;
}

__device__ inline void async_load_b128(unsigned lds, const void* gaddr) {
  asm volatile("global_load_async_to_lds_b128 %0, %1, off"
               :: "v"(lds), "v"(gaddr) : "memory");
}

__device__ inline void wait_async0() {
  asm volatile("s_wait_asynccnt 0x0" ::: "memory");
}

// ---------------------------------------------------------------------------
// Tiled f32 [K][N] -> f16 [N][K] transpose (weights, done once per launch).
// block (32,8), LDS 32x33 tile.
// ---------------------------------------------------------------------------
__global__ __launch_bounds__(256) void transpose_w_f16(const float* __restrict__ w,
                                                       _Float16* __restrict__ wt,
                                                       int K, int N) {
  __shared__ _Float16 t[32][33];
  int tx = threadIdx.x, ty = threadIdx.y;
  int n0 = blockIdx.x * 32, k0 = blockIdx.y * 32;
  #pragma unroll
  for (int i = 0; i < 4; ++i)
    t[ty*4 + i][tx] = (_Float16)w[(size_t)(k0 + ty*4 + i)*N + n0 + tx];
  __syncthreads();
  #pragma unroll
  for (int i = 0; i < 4; ++i)
    wt[(size_t)(n0 + ty*4 + i)*K + k0 + tx] = t[tx][ty*4 + i];
}

// ---------------------------------------------------------------------------
// V slice of qkv [token][3*EMBED] -> vt [b*h][d][key] (f16), so attention
// P*V B-fragments / LDS staging are contiguous 128-bit transfers.
// grid (SEQ/32, HDIM/32, B*HEADS), block (32,8)
// ---------------------------------------------------------------------------
__global__ __launch_bounds__(256) void transpose_v_f16(const _Float16* __restrict__ qkv,
                                                       _Float16* __restrict__ vt) {
  __shared__ _Float16 t[32][33];
  int tx = threadIdx.x, ty = threadIdx.y;
  int key0 = blockIdx.x * 32, d0 = blockIdx.y * 32, bh = blockIdx.z;
  int b = bh >> 4, h = bh & 15;
  #pragma unroll
  for (int i = 0; i < 4; ++i)
    t[ty*4 + i][tx] = qkv[(size_t)(b*SEQ + key0 + ty*4 + i)*(3*EMBED)
                          + 2*EMBED + h*HDIM + d0 + tx];
  __syncthreads();
  #pragma unroll
  for (int i = 0; i < 4; ++i)
    vt[(size_t)(bh*HDIM + d0 + ty*4 + i)*SEQ + key0 + tx] = t[tx][ty*4 + i];
}

// ---------------------------------------------------------------------------
// LayerNorm over rows of EMBED=1024, f32 in -> f16 out. One block per row.
// ---------------------------------------------------------------------------
__global__ __launch_bounds__(256) void layernorm_f16(const float* __restrict__ x,
                                                     const float* __restrict__ g,
                                                     const float* __restrict__ b,
                                                     _Float16* __restrict__ o) {
  __shared__ float s_[256], ss_[256];
  int row = blockIdx.x, tid = threadIdx.x;
  const float* xr = x + (size_t)row * EMBED;
  float4 v = *(const float4*)(xr + tid * 4);
  s_[tid]  = v.x + v.y + v.z + v.w;
  ss_[tid] = v.x*v.x + v.y*v.y + v.z*v.z + v.w*v.w;
  __syncthreads();
  for (int off = 128; off > 0; off >>= 1) {
    if (tid < off) { s_[tid] += s_[tid+off]; ss_[tid] += ss_[tid+off]; }
    __syncthreads();
  }
  float mu  = s_[0]  * (1.0f/EMBED);
  float var = ss_[0] * (1.0f/EMBED) - mu*mu;
  float inv = rsqrtf(var + 1e-6f);
  float4 gg = *(const float4*)(g + tid*4);
  float4 bb = *(const float4*)(b + tid*4);
  _Float16* orow = o + (size_t)row*EMBED + tid*4;
  orow[0] = (_Float16)((v.x-mu)*inv*gg.x + bb.x);
  orow[1] = (_Float16)((v.y-mu)*inv*gg.y + bb.y);
  orow[2] = (_Float16)((v.z-mu)*inv*gg.z + bb.z);
  orow[3] = (_Float16)((v.w-mu)*inv*gg.w + bb.w);
}

// ---------------------------------------------------------------------------
// Tiled WMMA GEMM: C[M,N] = A[M,K](f16) * Bt[N,K](f16, pre-transposed) + bias.
// Block tile 128x128x32, 8 waves (2Mx4N), wave tile 64x32 -> 4x2 wmma tiles.
// Double-buffered LDS staged with GLOBAL_LOAD_ASYNC_TO_LDS_B128 (ASYNCcnt):
//   wait(cur) -> barrier -> issue(next) -> WMMA(cur) -> barrier.
// ---------------------------------------------------------------------------
enum { EPI_F16 = 0, EPI_GELU = 1, EPI_RES = 2 };

template<int EPI>
__global__ __launch_bounds__(256) void gemm_wmma_f16(
    const _Float16* __restrict__ A, const _Float16* __restrict__ Bt,
    const float* __restrict__ bias, const float* __restrict__ res,
    _Float16* __restrict__ out16, float* __restrict__ out32,
    int M, int N, int K)
{
  constexpr int LDA = 40;  // padded row stride (16B-aligned: 80 bytes)
  __shared__ __align__(16) _Float16 As[2][128*LDA];
  __shared__ __align__(16) _Float16 Bs[2][128*LDA];

  int tid = threadIdx.x, lane = tid & 31, wid = tid >> 5;
  int l15 = lane & 15, hl = lane >> 4;
  int wm = (wid >> 2) * 64, wn = (wid & 3) * 32;
  int bM = blockIdx.y * 128, bN = blockIdx.x * 128;

  v8f zero = {0.f,0.f,0.f,0.f,0.f,0.f,0.f,0.f};
  v8f acc[4][2];
  #pragma unroll
  for (int i = 0; i < 4; ++i)
    #pragma unroll
    for (int j = 0; j < 2; ++j) acc[i][j] = zero;

  // staging map: thread covers row r (0..127), 32-byte chunk c ({0,16} halves)
  int r_ = tid >> 1, c_ = (tid & 1) * 16;
  const _Float16* Ag = A  + (size_t)(bM + r_) * K + c_;
  const _Float16* Bg = Bt + (size_t)(bN + r_) * K + c_;
  unsigned ldsA[2] = { lds_addr(&As[0][r_*LDA + c_]), lds_addr(&As[1][r_*LDA + c_]) };
  unsigned ldsB[2] = { lds_addr(&Bs[0][r_*LDA + c_]), lds_addr(&Bs[1][r_*LDA + c_]) };

  auto stage = [&](int buf, int kb) {
    async_load_b128(ldsA[buf],      Ag + kb);
    async_load_b128(ldsA[buf] + 16, Ag + kb + 8);
    async_load_b128(ldsB[buf],      Bg + kb);
    async_load_b128(ldsB[buf] + 16, Bg + kb + 8);
  };

  stage(0, 0);

  for (int kb = 0, it = 0; kb < K; kb += 32, ++it) {
    int cur = it & 1;
    wait_async0();
    __syncthreads();
    if (kb + 32 < K) stage(cur ^ 1, kb + 32);

    V16 af[4];
    #pragma unroll
    for (int tm = 0; tm < 4; ++tm) {               // A frag: ISA 16x32 layout
      int r = wm + tm*16 + l15;
      af[tm].h[0] = *(const v8h*)&As[cur][r*LDA + hl*8];
      af[tm].h[1] = *(const v8h*)&As[cur][r*LDA + hl*8 + 16];
    }
    #pragma unroll
    for (int tn = 0; tn < 2; ++tn) {               // B frag: 32x16 layout
      int n = wn + tn*16 + l15;
      V16 bf;
      bf.h[0] = *(const v8h*)&Bs[cur][n*LDA + hl*8];
      bf.h[1] = *(const v8h*)&Bs[cur][n*LDA + hl*8 + 16];
      #pragma unroll
      for (int tm = 0; tm < 4; ++tm)
        acc[tm][tn] = __builtin_amdgcn_wmma_f32_16x16x32_f16(
            false, af[tm].v, false, bf.v, (short)0, acc[tm][tn], false, false);
    }
    __syncthreads();   // all waves done with buf `cur` before it is re-staged
  }

  #pragma unroll
  for (int tm = 0; tm < 4; ++tm)
    #pragma unroll
    for (int tn = 0; tn < 2; ++tn) {
      int n = bN + wn + tn*16 + l15;
      float bv = bias[n];
      #pragma unroll
      for (int r = 0; r < 8; ++r) {
        int m = bM + wm + tm*16 + r + hl*8;        // C layout: M = r + 8*half
        float v = acc[tm][tn][r] + bv;
        size_t idx = (size_t)m * N + n;
        if constexpr (EPI == EPI_GELU) {
          v = 0.5f * v * (1.0f + erff(v * 0.70710678118654752f));
          out16[idx] = (_Float16)v;
        } else if constexpr (EPI == EPI_F16) {
          out16[idx] = (_Float16)v;
        } else {
          out32[idx] = res[idx] + v;
        }
      }
    }
}

// ---------------------------------------------------------------------------
// Flash attention: each block = 4 waves covering 64 q rows of ONE (b,h).
// Per 32-key iteration the block cooperatively stages the K block
// ([key][d], fragment-ready row-major) and V block ([d][key] from vt) into
// double-buffered LDS with one async b128 per thread each, overlapped with
// the WMMA work on the current buffer. Online softmax per wave; P converts
// C-layout -> A-layout through a wave-private LDS tile.
// grid: (B*HEADS) * (SEQ/64) blocks, 128 threads.
// ---------------------------------------------------------------------------
__global__ __launch_bounds__(128) void flash_attn_wmma(const _Float16* __restrict__ qkv,
                                                       const _Float16* __restrict__ vt,
                                                       _Float16* __restrict__ o) {
  constexpr int LDK = 72;   // K tile row stride ([key][d]), 144B, 16B-aligned
  constexpr int LDV = 40;   // V tile row stride ([d][key]), 80B, 16B-aligned
  __shared__ __align__(16) _Float16 Kb[2][32*LDK];
  __shared__ __align__(16) _Float16 Vb[2][64*LDV];
  __shared__ __align__(16) _Float16 Pb[4][16*32];

  int tid = threadIdx.x, lane = tid & 31, wid = tid >> 5;
  int bh = blockIdx.x >> 5;              // 32 (b,h) pairs
  int q0 = (blockIdx.x & 31) * 64;       // 64 q rows per block
  int bb = bh >> 4, hh = bh & 15;
  int l15 = lane & 15, hl = lane >> 4;
  _Float16* P_ = Pb[wid];

  // staging maps: one async b128 per thread per tile (2048 f16 per tile)
  int kKey = tid >> 2, kD = (tid & 3) * 16;   // K: 32 keys x 64 d
  int vD = tid >> 1, vKey = (tid & 1) * 16;   // V: 64 d x 32 keys
  const _Float16* kg = qkv + (size_t)(bb*SEQ + kKey)*(3*EMBED) + EMBED + hh*HDIM + kD;
  const _Float16* vg = vt + (size_t)(bh*HDIM + vD)*SEQ + vKey;
  unsigned ldsK[2] = { lds_addr(&Kb[0][kKey*LDK + kD]), lds_addr(&Kb[1][kKey*LDK + kD]) };
  unsigned ldsV[2] = { lds_addr(&Vb[0][vD*LDV + vKey]), lds_addr(&Vb[1][vD*LDV + vKey]) };

  auto stage = [&](int buf, int jb) {
    async_load_b128(ldsK[buf], kg + (size_t)jb*(3*EMBED));
    async_load_b128(ldsV[buf], vg + jb);
  };

  // Q fragments (A-matrix, 16x32 each, head_dim=64 -> 2 K-steps)
  const _Float16* qp = qkv + ((size_t)(bb*SEQ + q0 + wid*16 + l15)) * (3*EMBED) + hh*HDIM;
  V16 qf[2];
  #pragma unroll
  for (int kk = 0; kk < 2; ++kk) {
    qf[kk].h[0] = *(const v8h*)(qp + kk*32 + hl*8);
    qf[kk].h[1] = *(const v8h*)(qp + kk*32 + hl*8 + 16);
  }

  v8f zero = {0.f,0.f,0.f,0.f,0.f,0.f,0.f,0.f};
  v8f Of[4];
  #pragma unroll
  for (int t = 0; t < 4; ++t) Of[t] = zero;
  float Mr[8], Lr[8];
  #pragma unroll
  for (int r = 0; r < 8; ++r) { Mr[r] = -INFINITY; Lr[r] = 0.f; }
  const float scale = 0.125f;  // 1/sqrt(64)

  stage(0, 0);

  for (int j = 0; j < SEQ/32; ++j) {
    int cur = j & 1;
    int jb = j*32;
    wait_async0();
    __syncthreads();
    if (jb + 32 < SEQ) stage(cur ^ 1, jb + 32);

    // S = Q K^T for 32 keys (two 16x16 f32 tiles), K frags from LDS
    v8f s[2];
    #pragma unroll
    for (int tn = 0; tn < 2; ++tn) {
      s[tn] = zero;
      #pragma unroll
      for (int kk = 0; kk < 2; ++kk) {
        const _Float16* kp = &Kb[cur][(tn*16 + l15)*LDK + kk*32 + hl*8];
        V16 kf;
        kf.h[0] = *(const v8h*)kp;
        kf.h[1] = *(const v8h*)(kp + 16);
        s[tn] = __builtin_amdgcn_wmma_f32_16x16x32_f16(
            false, qf[kk].v, false, kf.v, (short)0, s[tn], false, false);
      }
      s[tn] = s[tn] * scale;
    }

    // online softmax (rows spread across 16-lane halves) + P to LDS
    #pragma unroll
    for (int r = 0; r < 8; ++r) {
      float a0 = s[0][r], a1 = s[1][r];
      float rm = fmaxf(a0, a1);
      #pragma unroll
      for (int off = 8; off >= 1; off >>= 1) rm = fmaxf(rm, __shfl_xor(rm, off, 32));
      float mn = fmaxf(Mr[r], rm);
      float alpha = __expf(Mr[r] - mn);
      float p0 = __expf(a0 - mn), p1 = __expf(a1 - mn);
      float rs = p0 + p1;
      #pragma unroll
      for (int off = 8; off >= 1; off >>= 1) rs += __shfl_xor(rs, off, 32);
      Lr[r] = Lr[r]*alpha + rs;
      Mr[r] = mn;
      #pragma unroll
      for (int t = 0; t < 4; ++t) Of[t][r] *= alpha;
      int mrow = r + hl*8;
      P_[mrow*32 + l15]      = (_Float16)p0;
      P_[mrow*32 + 16 + l15] = (_Float16)p1;
    }
    // wave-private P tile: same-wave DS ops complete in order, no barrier.

    // O += P(16x32) * V(32x64); V frags from LDS
    V16 pf;
    pf.h[0] = *(const v8h*)&P_[l15*32 + hl*8];
    pf.h[1] = *(const v8h*)&P_[l15*32 + hl*8 + 16];
    #pragma unroll
    for (int t = 0; t < 4; ++t) {
      const _Float16* vp = &Vb[cur][(t*16 + l15)*LDV + hl*8];
      V16 vf;
      vf.h[0] = *(const v8h*)vp;
      vf.h[1] = *(const v8h*)(vp + 16);
      Of[t] = __builtin_amdgcn_wmma_f32_16x16x32_f16(
          false, pf.v, false, vf.v, (short)0, Of[t], false, false);
    }
    __syncthreads();   // all waves done with buf `cur` before it is re-staged
  }

  // normalize and store (f16, ready as A for the proj GEMM)
  #pragma unroll
  for (int t = 0; t < 4; ++t)
    #pragma unroll
    for (int r = 0; r < 8; ++r) {
      int mrow = r + hl*8;
      size_t orow = (size_t)(bb*SEQ + q0 + wid*16 + mrow);
      o[orow*EMBED + hh*HDIM + t*16 + l15] = (_Float16)(Of[t][r] / Lr[r]);
    }
}

// ---------------------------------------------------------------------------
// Launch
// ---------------------------------------------------------------------------
extern "C" void kernel_launch(void* const* d_in, const int* in_sizes, int n_in,
                              void* d_out, int out_size, void* d_ws, size_t ws_size,
                              hipStream_t stream) {
  (void)in_sizes; (void)n_in; (void)out_size; (void)ws_size;
  const float* x      = (const float*)d_in[0];
  const float* ln1_g  = (const float*)d_in[1];
  const float* ln1_b  = (const float*)d_in[2];
  const float* w_qkv  = (const float*)d_in[3];
  const float* b_qkv  = (const float*)d_in[4];
  const float* w_proj = (const float*)d_in[5];
  const float* b_proj = (const float*)d_in[6];
  const float* ln2_g  = (const float*)d_in[7];
  const float* ln2_b  = (const float*)d_in[8];
  const float* w_fc1  = (const float*)d_in[9];
  const float* b_fc1  = (const float*)d_in[10];
  const float* w_fc2  = (const float*)d_in[11];
  const float* b_fc2  = (const float*)d_in[12];
  float* out = (float*)d_out;

  char* ws = (char*)d_ws;
  size_t off = 0;
  auto take = [&](size_t bytes) { char* p = ws + off; off += bytes; return (void*)p; };
  _Float16* w_qkv_t  = (_Float16*)take((size_t)EMBED*3*EMBED*2);   // [3E][E]
  _Float16* w_proj_t = (_Float16*)take((size_t)EMBED*EMBED*2);     // [E][E]
  _Float16* w_fc1_t  = (_Float16*)take((size_t)EMBED*HIDDEN*2);    // [H][E]
  _Float16* w_fc2_t  = (_Float16*)take((size_t)HIDDEN*EMBED*2);    // [E][H]
  _Float16* ln_h     = (_Float16*)take((size_t)TOKENS*EMBED*2);    // reused ln1/ln2
  _Float16* qkv_h    = (_Float16*)take((size_t)TOKENS*3*EMBED*2);
  _Float16* vt       = (_Float16*)take((size_t)TOKENS*EMBED*2);    // [b*h][d][key]
  _Float16* attno_h  = (_Float16*)take((size_t)TOKENS*EMBED*2);
  float*    x1       = (float*)   take((size_t)TOKENS*EMBED*4);
  _Float16* h_h      = (_Float16*)take((size_t)TOKENS*HIDDEN*2);

  dim3 tb(32, 8);
  // weights -> f16, transposed to [N][K]
  transpose_w_f16<<<dim3(3*EMBED/32, EMBED/32), tb, 0, stream>>>(w_qkv,  w_qkv_t,  EMBED, 3*EMBED);
  transpose_w_f16<<<dim3(EMBED/32,   EMBED/32), tb, 0, stream>>>(w_proj, w_proj_t, EMBED, EMBED);
  transpose_w_f16<<<dim3(HIDDEN/32,  EMBED/32), tb, 0, stream>>>(w_fc1,  w_fc1_t,  EMBED, HIDDEN);
  transpose_w_f16<<<dim3(EMBED/32,  HIDDEN/32), tb, 0, stream>>>(w_fc2,  w_fc2_t,  HIDDEN, EMBED);

  // ln1 -> qkv gemm -> (v transpose) -> attention -> proj (+residual x) -> x1
  layernorm_f16<<<TOKENS, 256, 0, stream>>>(x, ln1_g, ln1_b, ln_h);
  gemm_wmma_f16<EPI_F16><<<dim3(3*EMBED/128, TOKENS/128), 256, 0, stream>>>(
      ln_h, w_qkv_t, b_qkv, nullptr, qkv_h, nullptr, TOKENS, 3*EMBED, EMBED);
  transpose_v_f16<<<dim3(SEQ/32, HDIM/32, 2*HEADS), tb, 0, stream>>>(qkv_h, vt);
  flash_attn_wmma<<<(2*HEADS)*(SEQ/64), 128, 0, stream>>>(qkv_h, vt, attno_h);
  gemm_wmma_f16<EPI_RES><<<dim3(EMBED/128, TOKENS/128), 256, 0, stream>>>(
      attno_h, w_proj_t, b_proj, x, nullptr, x1, TOKENS, EMBED, EMBED);

  // ln2 -> fc1(+gelu) -> fc2 (+residual x1) -> out
  layernorm_f16<<<TOKENS, 256, 0, stream>>>(x1, ln2_g, ln2_b, ln_h);
  gemm_wmma_f16<EPI_GELU><<<dim3(HIDDEN/128, TOKENS/128), 256, 0, stream>>>(
      ln_h, w_fc1_t, b_fc1, nullptr, h_h, nullptr, TOKENS, HIDDEN, EMBED);
  gemm_wmma_f16<EPI_RES><<<dim3(EMBED/128, TOKENS/128), 256, 0, stream>>>(
      h_h, w_fc2_t, b_fc2, x1, nullptr, out, TOKENS, EMBED, HIDDEN);
}
